// NeuraMatch_36867999269287
// MI455X (gfx1250) — matrix operand order repaired and checked by reading, not compiled, not run on hardware
//
#include <hip/hip_runtime.h>
#include <math.h>

// ---------------------------------------------------------------------------
// NeuraMatch forward on MI455X (gfx1250), wave32 + WMMA + TDM.
//
// All convolutions (incl. transposed convs, expressed as pad=k-1 convs with
// flipped/transposed weights) and the pairwise matcher MLP run through
// v_wmma_f32_16x16x32_f16: f16 inputs, f32 accumulation.  Implicit im2col
// gathers A-fragments straight from global memory (everything fits in L2);
// weights are packed per call into a transposed f16 layout so a WMMA
// B-panel for one N-tile is a single contiguous block, which the Tensor
// Data Mover stages into LDS once per workgroup (4 waves share it).
// ---------------------------------------------------------------------------

typedef __attribute__((ext_vector_type(16))) _Float16 v16h;
typedef __attribute__((ext_vector_type(8)))  float    v8f;
typedef __attribute__((ext_vector_type(4)))  unsigned int u32x4;
typedef __attribute__((ext_vector_type(8)))  int      i32x8;
typedef __attribute__((ext_vector_type(4)))  int      i32x4;

#define SIDE   128
#define S2     16384      // 128*128
#define KTOP   768
#define PPAIR  294528LL   // 768*767/2
#define THRESH 0.5f

#define STR2(x) #x
#define STR(x) STR2(x)
#pragma message("probe: __clang_major__ = " STR(__clang_major__))

// ---------------------------------------------------------------------------
// Tensor Data Mover: 1-D contiguous copy of `nelem` 2-byte elements from
// global memory to LDS offset `lds_off`.  Descriptor per ISA 08_async_tensor
// §8: group0 {count=1, lds_addr, global_addr[56:0], type=2}, group1
// {data_size=1(2B), tensor_dim0=tile_dim0=nelem, tensor_dim1=1, stride0=nelem}.
// ---------------------------------------------------------------------------
#if __has_builtin(__builtin_amdgcn_tensor_load_to_lds)
#define HAVE_TDM 1
#pragma message("probe: HAVE_TDM = 1 (tensor_load_to_lds builtin available)")
__device__ __forceinline__ void tdm_load_1d(unsigned lds_off, const void* gptr,
                                            unsigned nelem) {
  unsigned long long ga = (unsigned long long)(uintptr_t)gptr;
  u32x4 g0;
  g0[0] = 1u;                                               // count=1, user mode
  g0[1] = lds_off;                                          // lds_addr (bytes)
  g0[2] = (unsigned)(ga & 0xFFFFFFFFu);                     // global_addr[31:0]
  g0[3] = (unsigned)((ga >> 32) & 0x1FFFFFFu) | (2u << 30); // addr[56:32] | type=2
  unsigned long long q0 = (1ull << 16)                      // data_size = 1 (2 bytes)
                        | ((unsigned long long)(nelem & 0xFFFFu) << 48); // tdim0 lo
  unsigned long long q1 = ((unsigned long long)(nelem >> 16) & 0xFFFFu)  // tdim0 hi
                        | (1ull << 16)                      // tensor_dim1 = 1
                        | ((unsigned long long)(nelem & 0xFFFFu) << 48); // tile_dim0
  unsigned long long q2 = ((unsigned long long)nelem) << 32; // tensor_dim0_stride
  unsigned long long q3 = 0;
  i32x8 g1;
  g1[0] = (int)q0; g1[1] = (int)(q0 >> 32);
  g1[2] = (int)q1; g1[3] = (int)(q1 >> 32);
  g1[4] = (int)q2; g1[5] = (int)(q2 >> 32);
  g1[6] = (int)q3; g1[7] = (int)(q3 >> 32);
  i32x4 z4 = {0, 0, 0, 0};
#if defined(__clang_major__) && (__clang_major__ >= 23)
  i32x8 z8 = {0, 0, 0, 0, 0, 0, 0, 0};
  __builtin_amdgcn_tensor_load_to_lds(g0, g1, z4, z4, z8, 0);
#else
  __builtin_amdgcn_tensor_load_to_lds(g0, g1, z4, z4, 0);
#endif
}
#else
#define HAVE_TDM 0
#pragma message("probe: HAVE_TDM = 0 (tensor_load_to_lds builtin NOT available)")
#endif

// ---------------------------------------------------------------------------
// Weight packing: produce Bt[NPad][KdPad] (f16), i.e. column-of-B major so a
// WMMA B-fragment (16 consecutive K values for one output channel) is one
// contiguous 32-byte run, and a whole 16-column N-panel is contiguous.
//   mode 0: conv    W[o][c][r][s] = w[o][c][r][s]              (w: cout,cin,k,k)
//   mode 1: convT   W[o][c][r][s] = w[c][o][k-1-r][k-1-s]      (w: cin,cout,k,k)
//   mode 2: linear  W[o][c]       = w[c][o]                    (w: din,dout), k==1
// ---------------------------------------------------------------------------
__global__ void pack_w_kernel(const float* __restrict__ w, _Float16* __restrict__ Bt,
                              int cout, int cin, int k, int mode, int KdPad, int NPad) {
  long long i = (long long)blockIdx.x * blockDim.x + threadIdx.x;
  long long total = (long long)NPad * KdPad;
  if (i >= total) return;
  int o  = (int)(i / KdPad);
  int kd = (int)(i % KdPad);
  int kk2 = k * k;
  float v = 0.0f;
  if (o < cout && kd < cin * kk2) {
    int c  = kd / kk2;
    int rs = kd - c * kk2;
    int r  = rs / k;
    int s  = rs - r * k;
    if (mode == 0)      v = w[(((long long)o * cin + c) * k + r) * k + s];
    else if (mode == 1) v = w[(((long long)c * cout + o) * k + (k - 1 - r)) * k + (k - 1 - s)];
    else                v = w[(long long)c * cout + o];
  }
  Bt[i] = (_Float16)v;
}

__global__ void zero_kernel(float* p, int n) {
  int i = blockIdx.x * blockDim.x + threadIdx.x;
  if (i < n) p[i] = 0.0f;
}

// ---------------------------------------------------------------------------
// Generic implicit-im2col WMMA conv.  Input = concat of up to 3 NCHW views.
// Block = 128 threads = 4 waves; each wave computes one 16(M) x 16(N) tile,
// all four share one B panel ([16][KdPad] f16) staged in LDS by the TDM.
// A colmap LDS table caches the col -> (c,r,s) im2col decode.
// Epilogue mode 0: BN + LeakyReLU(0.01).  mode 1: sigmoid(acc + bias).
// ---------------------------------------------------------------------------
struct ConvDesc {
  const float* in0; const float* in1; const float* in2;
  int c0, c1, c2;
  long long bs0, bs1, bs2;         // batch strides (elements)
  int IH, IW, OH, OW, ksz, pad, cin, cout;
  const _Float16* Bt;              // [NPad][KdPad]
  int KdPad, NPad;
  int mode;                        // epilogue
  const float* gg; const float* bb; const float* mm; const float* vv;
  const float* bias;
  float* out;                      // (2, cout, OH, OW)
};

extern __shared__ char dyn_smem[];

__global__ __launch_bounds__(128) void conv_wmma_kernel(ConvDesc d) {
  _Float16* panel  = (_Float16*)dyn_smem;                       // [16][KdPad]
  int*      colmap = (int*)(dyn_smem + (size_t)d.KdPad * 32);   // [KdPad]

  const int tid  = threadIdx.x;
  const int wave = tid >> 5;
  const int lane = tid & 31;
  const int tm = blockIdx.x * 4 + wave;       // this wave's M tile
  const int tn = blockIdx.y;                  // N tile (shared across waves)
  const int hw = d.OH * d.OW;
  const int M  = 2 * hw;
  const int kk2 = d.ksz * d.ksz;
  const int KK  = d.cin * kk2;

  // ---- build im2col decode table (once per block) ----
  for (int col = tid; col < d.KdPad; col += 128) {
    int packed = -1;
    if (col < KK) {
      int c  = col / kk2;
      int rs = col - c * kk2;
      int r  = rs / d.ksz;
      int s  = rs - r * d.ksz;
      packed = (c << 10) | (r << 5) | s;      // c:21b | r:5b | s:5b
    }
    colmap[col] = packed;
  }

  // ---- stage B panel: rows [tn*16, tn*16+16) of Bt, contiguous ----
#if HAVE_TDM
  if (wave == 0) {
    tdm_load_1d(0u, d.Bt + (size_t)tn * 16 * d.KdPad, (unsigned)(16 * d.KdPad));
    __builtin_amdgcn_s_wait_tensorcnt(0);
  }
#else
  {
    const _Float16* src = d.Bt + (size_t)tn * 16 * d.KdPad;
    for (int i = tid; i < 16 * d.KdPad; i += 128) panel[i] = src[i];
  }
#endif
  __syncthreads();

  // A-fragment row for this lane (ISA 7.12.2: lanes 0-15 and 16-31 both M=0..15)
  const int arow = tm * 16 + (lane & 15);
  int an   = arow / hw;
  int arem = arow - an * hw;
  int aoy  = arem / d.OW;
  int aox  = arem - aoy * d.OW;
  const bool avalid = (arow < M);
  const int khalf = (lane < 16) ? 0 : 16;

  // hint the next input rows into cache (global_prefetch_b8)
  if (avalid && d.in0) __builtin_prefetch(d.in0 + (long long)an * d.bs0 + (long long)aoy * d.IW, 0, 3);

  // B-fragment source in LDS: panel[row=lane&15][kd]
  const _Float16* bbase = panel + (size_t)(lane & 15) * d.KdPad + ((lane < 16) ? 0 : 16);

  v8f acc = {};
  for (int kk = 0; kk < d.KdPad; kk += 32) {
    v16h a;
#pragma unroll
    for (int j = 0; j < 16; ++j) {
      int koff = j + khalf;   // K offset within 32-slice
      int pk = colmap[kk + koff];
      float v = 0.0f;
      if (avalid && pk >= 0) {
        int c = pk >> 10;
        int r = (pk >> 5) & 31;
        int s = pk & 31;
        int iy = aoy + r - d.pad;
        int ix = aox + s - d.pad;
        if (iy >= 0 && iy < d.IH && ix >= 0 && ix < d.IW) {
          const float* src; long long off;
          if (c < d.c0) {
            src = d.in0; off = (long long)an * d.bs0 + ((long long)c * d.IH + iy) * d.IW + ix;
          } else if (c < d.c0 + d.c1) {
            src = d.in1; off = (long long)an * d.bs1 + ((long long)(c - d.c0) * d.IH + iy) * d.IW + ix;
          } else {
            src = d.in2; off = (long long)an * d.bs2 + ((long long)(c - d.c0 - d.c1) * d.IH + iy) * d.IW + ix;
          }
          v = src[off];
        }
      }
      a[j] = (_Float16)v;
    }
    const v16h b = *(const v16h*)(bbase + kk);    // 32B, 32B-aligned (KdPad%32==0)

    acc = __builtin_amdgcn_wmma_f32_16x16x32_f16(false, a, false, b, (short)0, acc,
                                                 false, false);
  }

  // Epilogue (C/D layout: lanes 0-15 -> rows 0..7, lanes 16-31 -> rows 8..15)
  const int ncol = tn * 16 + (lane & 15);
  if (ncol >= d.cout) return;
  float g_ = 0.f, b_ = 0.f, m_ = 0.f, vv_ = 0.f, bias_ = 0.f;
  if (d.mode == 0) { g_ = d.gg[ncol]; b_ = d.bb[ncol]; m_ = d.mm[ncol]; vv_ = d.vv[ncol]; }
  else             { bias_ = d.bias[ncol]; }
  const float rstd = (d.mode == 0) ? rsqrtf(vv_ + 1e-5f) : 0.f;
#pragma unroll
  for (int r = 0; r < 8; ++r) {
    int mrow = tm * 16 + r + ((lane < 16) ? 0 : 8);
    if (mrow >= M) continue;
    int n   = mrow / hw;
    int rem = mrow - n * hw;
    int oy  = rem / d.OW;
    int ox  = rem - oy * d.OW;
    float val = acc[r];
    if (d.mode == 0) {
      val = (val - m_) * g_ * rstd + b_;
      val = (val > 0.f) ? val : 0.01f * val;
    } else {
      val = 1.0f / (1.0f + __expf(-(val + bias_)));
    }
    d.out[(((long long)n * d.cout + ncol) * d.OH + oy) * d.OW + ox] = val;
  }
}

// ---------------------------------------------------------------------------
// Top-K(768) of a 16384-element heatmap channel, matching jax.lax.top_k
// semantics on masked input (v>0.5 ? v : -inf): descending values, ties by
// ascending index.  Deterministic iterative argmax; heat stays hot in L0/L2.
// key: removed=0 < masked=1 < float bits of v (all v in (0.5,1], positive).
// ---------------------------------------------------------------------------
__global__ __launch_bounds__(256) void topk_kernel(const float* __restrict__ heat,
                                                   int* __restrict__ idx_out) {
  __shared__ unsigned int flag[512];   // 16384 removal bits
  __shared__ unsigned int rk[256];
  __shared__ int          rix[256];
  const int tid = threadIdx.x;
  const int bc  = blockIdx.x;                  // b*2 + channel
  const float* hm = heat + (long long)bc * S2;

  for (int i = tid; i < 512; i += 256) flag[i] = 0u;
  __syncthreads();

  for (int r = 0; r < KTOP; ++r) {
    unsigned int bestk = 0u; int besti = 0x7FFFFFFF;
    for (int i = tid; i < S2; i += 256) {
      if (flag[i >> 5] & (1u << (i & 31))) continue;
      float v = hm[i];
      unsigned int key = (v > THRESH) ? __float_as_uint(v) : 1u;
      if (key > bestk || (key == bestk && i < besti)) { bestk = key; besti = i; }
    }
    rk[tid] = bestk; rix[tid] = besti;
    __syncthreads();
    for (int s = 128; s > 0; s >>= 1) {
      if (tid < s) {
        unsigned int k2 = rk[tid + s]; int i2 = rix[tid + s];
        if (k2 > rk[tid] || (k2 == rk[tid] && i2 < rix[tid])) { rk[tid] = k2; rix[tid] = i2; }
      }
      __syncthreads();
    }
    if (tid == 0) {
      int bi = rix[0];
      idx_out[(long long)bc * KTOP + r] = bi;
      flag[bi >> 5] |= (1u << (bi & 31));
    }
    __syncthreads();
  }
}

// ---------------------------------------------------------------------------
// Matcher: one wave per 16 pairs.  Three fused WMMA layers
// (64->32->16->1, K/N padded to WMMA shapes), LDS relayout between layers.
// y = (sigmoid(h3)+hmA[pa]+hmB[pb])/3; scatter-max into matched_conf via
// atomicMax on float bits (all candidates positive).
// ---------------------------------------------------------------------------
__global__ __launch_bounds__(32) void match_kernel(
    const float* __restrict__ fa, const float* __restrict__ fb,   // (2,32,S2)
    const float* __restrict__ heat,                               // (2,2,S2)
    const int*   __restrict__ idx,                                // (2,2,768)
    const _Float16* __restrict__ w1t,   // [32][64]
    const _Float16* __restrict__ w2t,   // [16][32]
    const _Float16* __restrict__ w3t,   // [16][32] (only col 0 / K<16 nonzero)
    const float* __restrict__ b1, const float* __restrict__ b2,
    const float* __restrict__ b3,
    float* __restrict__ y_all, float* __restrict__ conf) {
  const int lane = threadIdx.x;
  const long long p0 = (long long)blockIdx.x * 16;
  const int b = blockIdx.y;

  __shared__ int   s_pa[16], s_pb[16], s_q[16];
  __shared__ float s_ya[16], s_yb[16];
  __shared__ _Float16 s_h[16][32];

  if (lane < 16) {
    long long p = p0 + lane;
    // invert p -> (q, r) for triu_indices(768, k=1); cum(q) = q*(1535-q)/2
    float disc = 1535.0f * 1535.0f - 8.0f * (float)p;
    int q = (int)floorf((1535.0f - sqrtf(disc)) * 0.5f);
    if (q < 0) { q = 0; }
    if (q > 766) { q = 766; }
    while (q > 0 && (long long)q * (1535 - q) / 2 > p) --q;
    while (q < 766 && (long long)(q + 1) * (1535 - (q + 1)) / 2 <= p) ++q;
    long long base = (long long)q * (1535 - q) / 2;
    int r = (int)(p - base) + q + 1;
    int pa = idx[((long long)b * 2 + 0) * KTOP + q];
    int pb = idx[((long long)b * 2 + 1) * KTOP + r];
    s_q[lane] = q; s_pa[lane] = pa; s_pb[lane] = pb;
    s_ya[lane] = heat[((long long)b * 2 + 0) * S2 + pa];
    s_yb[lane] = heat[((long long)b * 2 + 1) * S2 + pb];
  }
  __syncthreads();

  // ---- layer 1: [16x64] @ [64x32] ----
  v8f acc0 = {}, acc1 = {};
  {
    const int arow = lane & 15;
    const int pa = s_pa[arow], pb = s_pb[arow];
    const int khalf = (lane < 16) ? 0 : 16;
    const int c0 = lane & 15;
    const _Float16* pw0 = w1t + (long long)c0 * 64 + ((lane < 16) ? 0 : 16);
    const _Float16* pw1 = w1t + (long long)(16 + c0) * 64 + ((lane < 16) ? 0 : 16);
#pragma unroll
    for (int kk = 0; kk < 64; kk += 32) {
      v16h a;
#pragma unroll
      for (int j = 0; j < 16; ++j) {
        int koff = j + khalf;
        int col = kk + koff;
        float v = (col < 32)
            ? fa[((long long)b * 32 + col) * S2 + pa]
            : fb[((long long)b * 32 + (col - 32)) * S2 + pb];
        a[j] = (_Float16)v;
      }
      v16h bf0, bf1;
#pragma unroll
      for (int j = 0; j < 16; ++j) { bf0[j] = pw0[kk + j]; bf1[j] = pw1[kk + j]; }
      acc0 = __builtin_amdgcn_wmma_f32_16x16x32_f16(false, a, false, bf0, (short)0, acc0, false, false);
      acc1 = __builtin_amdgcn_wmma_f32_16x16x32_f16(false, a, false, bf1, (short)0, acc1, false, false);
    }
  }
  {
    int n0 = lane & 15;
    float bb0 = b1[n0], bb1 = b1[16 + n0];
#pragma unroll
    for (int r = 0; r < 8; ++r) {
      int mrow = r + ((lane < 16) ? 0 : 8);
      float v0 = acc0[r] + bb0; v0 = (v0 > 0.f) ? v0 : 0.01f * v0;
      float v1 = acc1[r] + bb1; v1 = (v1 > 0.f) ? v1 : 0.01f * v1;
      s_h[mrow][n0] = (_Float16)v0;
      s_h[mrow][16 + n0] = (_Float16)v1;
    }
  }
  __syncthreads();

  // ---- layer 2: [16x32] @ [32x16] ----
  v16h a2, bf2;
  {
    const int khalf = (lane < 16) ? 0 : 16;
#pragma unroll
    for (int j = 0; j < 16; ++j) {
      int koff = j + khalf;
      a2[j] = s_h[lane & 15][koff];
    }
    const _Float16* pw = w2t + (long long)(lane & 15) * 32 + ((lane < 16) ? 0 : 16);
#pragma unroll
    for (int j = 0; j < 16; ++j) bf2[j] = pw[j];
  }
  v8f acc2 = {};
  acc2 = __builtin_amdgcn_wmma_f32_16x16x32_f16(false, a2, false, bf2, (short)0, acc2, false, false);
  __syncthreads();
  {
    int n0 = lane & 15;
    float bb = b2[n0];
#pragma unroll
    for (int r = 0; r < 8; ++r) {
      int mrow = r + ((lane < 16) ? 0 : 8);
      float v = acc2[r] + bb; v = (v > 0.f) ? v : 0.01f * v;
      s_h[mrow][n0] = (_Float16)v;
      s_h[mrow][16 + n0] = (_Float16)0.f;     // K pad for layer 3
    }
  }
  __syncthreads();

  // ---- layer 3: [16x32] @ [32x16] (only column 0 meaningful) ----
  v16h a3, bf3;
  {
    const int khalf = (lane < 16) ? 0 : 16;
#pragma unroll
    for (int j = 0; j < 16; ++j) {
      int koff = j + khalf;
      a3[j] = s_h[lane & 15][koff];
    }
    const _Float16* pw = w3t + (long long)(lane & 15) * 32 + ((lane < 16) ? 0 : 16);
#pragma unroll
    for (int j = 0; j < 16; ++j) bf3[j] = pw[j];
  }
  v8f acc3 = {};
  acc3 = __builtin_amdgcn_wmma_f32_16x16x32_f16(false, a3, false, bf3, (short)0, acc3, false, false);

  if ((lane & 15) == 0) {          // lanes 0 (rows 0-7) and 16 (rows 8-15) hold column 0
    float bb = b3[0];
#pragma unroll
    for (int r = 0; r < 8; ++r) {
      int mrow = r + ((lane < 16) ? 0 : 8);
      long long p = p0 + mrow;
      float m = 1.0f / (1.0f + __expf(-(acc3[r] + bb)));
      float y = (m + s_ya[mrow] + s_yb[mrow]) * (1.0f / 3.0f);
      y_all[(long long)b * PPAIR + p] = y;
      if (y > THRESH) {
        atomicMax((unsigned int*)&conf[(long long)b * KTOP + s_q[mrow]], __float_as_uint(y));
      }
    }
  }
}

// ---------------------------------------------------------------------------
// Host orchestration
// ---------------------------------------------------------------------------
struct View { const float* p; int c; long long bs; };

static void conv_layer(hipStream_t st, View v0, View v1, View v2,
                       int IH, int IW, int k, int pad, int cout,
                       const float* w, int wmode, _Float16* Bt,
                       int emode, const float* g, const float* b,
                       const float* m, const float* vv, const float* bias,
                       float* out, int OH, int OW) {
  int cin = v0.c + v1.c + v2.c;
  int KdPad = (cin * k * k + 31) / 32 * 32;
  int NPad  = (cout + 15) / 16 * 16;
  long long tot = (long long)KdPad * NPad;
  pack_w_kernel<<<dim3((unsigned)((tot + 255) / 256)), 256, 0, st>>>(
      w, Bt, cout, cin, k, wmode, KdPad, NPad);
  ConvDesc d;
  d.in0 = v0.p; d.in1 = v1.p; d.in2 = v2.p;
  d.c0 = v0.c; d.c1 = v1.c; d.c2 = v2.c;
  d.bs0 = v0.bs; d.bs1 = v1.bs; d.bs2 = v2.bs;
  d.IH = IH; d.IW = IW; d.OH = OH; d.OW = OW; d.ksz = k; d.pad = pad;
  d.cin = cin; d.cout = cout;
  d.Bt = Bt; d.KdPad = KdPad; d.NPad = NPad; d.mode = emode;
  d.gg = g; d.bb = b; d.mm = m; d.vv = vv; d.bias = bias; d.out = out;
  int M = 2 * OH * OW;
  dim3 grid((M + 63) / 64, NPad / 16);
  size_t shmem = (size_t)KdPad * 32 + (size_t)KdPad * 4;   // B panel + colmap
  conv_wmma_kernel<<<grid, 128, shmem, st>>>(d);
}

// Flattened input order: assumes JAX pytree flattening (dict keys sorted
// alphabetically, lists in order).  Top level: params < x.  Per layer dict:
// b, g, m, v, w.  cab_out/hc_out: b, w.  matcher: b1,b2,b3,w1,w2,w3.
enum {
  CA0_B = 0, CA0_G, CA0_M, CA0_V, CA0_W,
  CA1_B, CA1_G, CA1_M, CA1_V, CA1_W,
  CA2_B, CA2_G, CA2_M, CA2_V, CA2_W,
  CAB0_B, CAB0_G, CAB0_M, CAB0_V, CAB0_W,
  CAB1_B, CAB1_G, CAB1_M, CAB1_V, CAB1_W,
  CAB2_B, CAB2_G, CAB2_M, CAB2_V, CAB2_W,
  CAB3_B, CAB3_G, CAB3_M, CAB3_V, CAB3_W,
  CABOUT_B, CABOUT_W,
  CB0_B, CB0_G, CB0_M, CB0_V, CB0_W,
  CB1_B, CB1_G, CB1_M, CB1_V, CB1_W,
  CB2_B, CB2_G, CB2_M, CB2_V, CB2_W,
  DESC0_B, DESC0_G, DESC0_M, DESC0_V, DESC0_W,
  DESC1_B, DESC1_G, DESC1_M, DESC1_V, DESC1_W,
  HC0_B, HC0_G, HC0_M, HC0_V, HC0_W,
  HCOUT_B, HCOUT_W,
  MAT_B1, MAT_B2, MAT_B3, MAT_W1, MAT_W2, MAT_W3,
  X_IN
};

extern "C" void kernel_launch(void* const* d_in, const int* in_sizes, int n_in,
                              void* d_out, int out_size, void* d_ws, size_t ws_size,
                              hipStream_t stream) {
  (void)in_sizes; (void)n_in; (void)out_size; (void)ws_size;
  auto F = [&](int i) { return (const float*)d_in[i]; };
  const float* x = F(X_IN);

  // output regions: (heatmap, y_all, matched_conf) concatenated
  float* heatmap = (float*)d_out;                 // 2*2*128*128 = 65536
  float* y_all   = heatmap + 65536;               // 2*294528  = 589056
  float* conf    = y_all + 589056;                // 2*768     = 1536

  // workspace bump allocator
  char* base = (char*)d_ws; size_t off = 0;
  auto AF = [&](size_t n) { float* p = (float*)(base + off);
                            off += (n * sizeof(float) + 255) / 256 * 256; return p; };
  auto AH = [&](size_t n) { _Float16* p = (_Float16*)(base + off);
                            off += (n * sizeof(_Float16) + 255) / 256 * 256; return p; };
  auto AI = [&](size_t n) { int* p = (int*)(base + off);
                            off += (n * sizeof(int) + 255) / 256 * 256; return p; };

  // activations (f32)
  float* a1     = AF((size_t)2 * 64 * 123 * 123);   // shared by ca/cb chains
  float* a2     = AF((size_t)2 * 64 * 120 * 120);
  float* fa_raw = AF((size_t)2 * 32 * 118 * 118);
  float* fb_raw = AF((size_t)2 * 32 * 118 * 118);
  float* h1     = AF((size_t)2 * 64 * 126 * 126);
  float* h2     = AF((size_t)2 * 64 * 124 * 124);
  float* h3     = AF((size_t)2 * 128 * 122 * 122);
  float* h4     = AF((size_t)2 * 32 * 120 * 120);
  float* hmraw  = AF((size_t)2 * 2 * 118 * 118);
  float* hc1    = AF((size_t)2 * 16 * 123 * 123);
  float* fa_t   = AF((size_t)2 * 32 * 123 * 123);
  float* fb_t   = AF((size_t)2 * 32 * 123 * 123);
  float* fa     = AF((size_t)2 * 32 * 128 * 128);
  float* fb     = AF((size_t)2 * 32 * 128 * 128);
  int*   topidx = AI((size_t)2 * 2 * KTOP);

#define KDP(cin, k) ((((cin) * (k) * (k)) + 31) / 32 * 32)
#define NP(cout)    (((cout) + 15) / 16 * 16)
  // packed weights (f16, transposed)
  _Float16* bt_ca0   = AH((size_t)KDP(3, 6)   * NP(64));
  _Float16* bt_ca1   = AH((size_t)KDP(64, 4)  * NP(64));
  _Float16* bt_ca2   = AH((size_t)KDP(64, 3)  * NP(32));
  _Float16* bt_cb0   = AH((size_t)KDP(3, 6)   * NP(64));
  _Float16* bt_cb1   = AH((size_t)KDP(64, 4)  * NP(64));
  _Float16* bt_cb2   = AH((size_t)KDP(64, 3)  * NP(32));
  _Float16* bt_cab0  = AH((size_t)KDP(6, 3)   * NP(64));
  _Float16* bt_cab1  = AH((size_t)KDP(64, 3)  * NP(64));
  _Float16* bt_cab2  = AH((size_t)KDP(64, 3)  * NP(128));
  _Float16* bt_cab3  = AH((size_t)KDP(128, 3) * NP(32));
  _Float16* bt_cabo  = AH((size_t)KDP(32, 3)  * NP(2));
  _Float16* bt_hc    = AH((size_t)KDP(66, 6)  * NP(16));
  _Float16* bt_hco   = AH((size_t)KDP(16, 6)  * NP(2));
  _Float16* bt_d0a   = AH((size_t)KDP(32, 6)  * NP(32));
  _Float16* bt_d1a   = AH((size_t)KDP(32, 6)  * NP(32));
  _Float16* bt_d0b   = AH((size_t)KDP(32, 6)  * NP(32));
  _Float16* bt_d1b   = AH((size_t)KDP(32, 6)  * NP(32));
  _Float16* w1t      = AH((size_t)64 * 32);
  _Float16* w2t      = AH((size_t)32 * 16);
  _Float16* w3t      = AH((size_t)32 * 16);

  View VN{nullptr, 0, 0};
  View vxa{x, 3, (long long)2 * 3 * S2};              // x[:,0]
  View vxb{x + 3 * S2, 3, (long long)2 * 3 * S2};     // x[:,1]

  // ---- ca chain: x_a -> fa_raw ----
  conv_layer(stream, vxa, VN, VN, 128, 128, 6, 0, 64, F(CA0_W), 0, bt_ca0,
             0, F(CA0_G), F(CA0_B), F(CA0_M), F(CA0_V), nullptr, a1, 123, 123);
  conv_layer(stream, View{a1, 64, (long long)64 * 123 * 123}, VN, VN, 123, 123, 4, 0, 64,
             F(CA1_W), 0, bt_ca1, 0, F(CA1_G), F(CA1_B), F(CA1_M), F(CA1_V), nullptr, a2, 120, 120);
  conv_layer(stream, View{a2, 64, (long long)64 * 120 * 120}, VN, VN, 120, 120, 3, 0, 32,
             F(CA2_W), 0, bt_ca2, 0, F(CA2_G), F(CA2_B), F(CA2_M), F(CA2_V), nullptr, fa_raw, 118, 118);

  // ---- cb chain: x_b -> fb_raw (reuses a1/a2) ----
  conv_layer(stream, vxb, VN, VN, 128, 128, 6, 0, 64, F(CB0_W), 0, bt_cb0,
             0, F(CB0_G), F(CB0_B), F(CB0_M), F(CB0_V), nullptr, a1, 123, 123);
  conv_layer(stream, View{a1, 64, (long long)64 * 123 * 123}, VN, VN, 123, 123, 4, 0, 64,
             F(CB1_W), 0, bt_cb1, 0, F(CB1_G), F(CB1_B), F(CB1_M), F(CB1_V), nullptr, a2, 120, 120);
  conv_layer(stream, View{a2, 64, (long long)64 * 120 * 120}, VN, VN, 120, 120, 3, 0, 32,
             F(CB2_W), 0, bt_cb2, 0, F(CB2_G), F(CB2_B), F(CB2_M), F(CB2_V), nullptr, fb_raw, 118, 118);

  // ---- cab chain: concat(x_a, x_b) -> h4 -> heatmap_raw ----
  conv_layer(stream, vxa, vxb, VN, 128, 128, 3, 0, 64, F(CAB0_W), 0, bt_cab0,
             0, F(CAB0_G), F(CAB0_B), F(CAB0_M), F(CAB0_V), nullptr, h1, 126, 126);
  conv_layer(stream, View{h1, 64, (long long)64 * 126 * 126}, VN, VN, 126, 126, 3, 0, 64,
             F(CAB1_W), 0, bt_cab1, 0, F(CAB1_G), F(CAB1_B), F(CAB1_M), F(CAB1_V), nullptr, h2, 124, 124);
  conv_layer(stream, View{h2, 64, (long long)64 * 124 * 124}, VN, VN, 124, 124, 3, 0, 128,
             F(CAB2_W), 0, bt_cab2, 0, F(CAB2_G), F(CAB2_B), F(CAB2_M), F(CAB2_V), nullptr, h3, 122, 122);
  conv_layer(stream, View{h3, 128, (long long)128 * 122 * 122}, VN, VN, 122, 122, 3, 0, 32,
             F(CAB3_W), 0, bt_cab3, 0, F(CAB3_G), F(CAB3_B), F(CAB3_M), F(CAB3_V), nullptr, h4, 120, 120);
  conv_layer(stream, View{h4, 32, (long long)32 * 120 * 120}, VN, VN, 120, 120, 3, 0, 2,
             F(CABOUT_W), 0, bt_cabo, 1, nullptr, nullptr, nullptr, nullptr, F(CABOUT_B),
             hmraw, 118, 118);

  // ---- hc: concat(fa_raw, fb_raw, heatmap_raw) -> hc1 -> heatmap (d_out) ----
  conv_layer(stream,
             View{fa_raw, 32, (long long)32 * 118 * 118},
             View{fb_raw, 32, (long long)32 * 118 * 118},
             View{hmraw,  2,  (long long)2  * 118 * 118},
             118, 118, 6, 5, 16, F(HC0_W), 1, bt_hc,
             0, F(HC0_G), F(HC0_B), F(HC0_M), F(HC0_V), nullptr, hc1, 123, 123);
  conv_layer(stream, View{hc1, 16, (long long)16 * 123 * 123}, VN, VN, 123, 123, 6, 5, 2,
             F(HCOUT_W), 1, bt_hco, 1, nullptr, nullptr, nullptr, nullptr, F(HCOUT_B),
             heatmap, 128, 128);

  // ---- desc chains: fa_raw/fb_raw -> fa/fb (convT x2, shared weights) ----
  conv_layer(stream, View{fa_raw, 32, (long long)32 * 118 * 118}, VN, VN, 118, 118, 6, 5, 32,
             F(DESC0_W), 1, bt_d0a, 0, F(DESC0_G), F(DESC0_B), F(DESC0_M), F(DESC0_V), nullptr,
             fa_t, 123, 123);
  conv_layer(stream, View{fa_t, 32, (long long)32 * 123 * 123}, VN, VN, 123, 123, 6, 5, 32,
             F(DESC1_W), 1, bt_d1a, 0, F(DESC1_G), F(DESC1_B), F(DESC1_M), F(DESC1_V), nullptr,
             fa, 128, 128);
  conv_layer(stream, View{fb_raw, 32, (long long)32 * 118 * 118}, VN, VN, 118, 118, 6, 5, 32,
             F(DESC0_W), 1, bt_d0b, 0, F(DESC0_G), F(DESC0_B), F(DESC0_M), F(DESC0_V), nullptr,
             fb_t, 123, 123);
  conv_layer(stream, View{fb_t, 32, (long long)32 * 123 * 123}, VN, VN, 123, 123, 6, 5, 32,
             F(DESC1_W), 1, bt_d1b, 0, F(DESC1_G), F(DESC1_B), F(DESC1_M), F(DESC1_V), nullptr,
             fb, 128, 128);

  // ---- matcher weight packing (linear mode) ----
  pack_w_kernel<<<dim3((64 * 32 + 255) / 256), 256, 0, stream>>>(F(MAT_W1), w1t, 32, 64, 1, 2, 64, 32);
  pack_w_kernel<<<dim3((32 * 16 + 255) / 256), 256, 0, stream>>>(F(MAT_W2), w2t, 16, 32, 1, 2, 32, 16);
  pack_w_kernel<<<dim3((32 * 16 + 255) / 256), 256, 0, stream>>>(F(MAT_W3), w3t, 1, 16, 1, 2, 32, 16);

  // ---- top-k + matcher ----
  zero_kernel<<<dim3((1536 + 255) / 256), 256, 0, stream>>>(conf, 1536);
  topk_kernel<<<dim3(4), 256, 0, stream>>>(heatmap, topidx);
  match_kernel<<<dim3((unsigned)(PPAIR / 16), 2), 32, 0, stream>>>(
      fa, fb, heatmap, topidx, w1t, w2t, w3t,
      F(MAT_B1), F(MAT_B2), F(MAT_B3), y_all, conf);
#undef KDP
#undef NP
}
